// TemporalTransformer_90288802497062
// MI455X (gfx1250) — compile-verified, matrix operand order checked by reading, not run on hardware
//
#include <hip/hip_runtime.h>
#include <math.h>

typedef unsigned short u16;
typedef __attribute__((ext_vector_type(16))) __bf16 v16bf;
typedef __attribute__((ext_vector_type(8)))  float  v8f;
typedef unsigned int  u32x4 __attribute__((ext_vector_type(4)));
typedef int           i32x8 __attribute__((ext_vector_type(8)));
typedef int           i32x4 __attribute__((ext_vector_type(4)));

#define BB 4
#define TT 512
#define DIMM 768
#define HEADSN 12
#define DHH 64
#define INNERN 768
#define FFQ 3072
#define DEPTHN 6
#define MROWS (BB*TT)

#if defined(__has_builtin)
#if __has_builtin(__builtin_amdgcn_tensor_load_to_lds) && __has_builtin(__builtin_amdgcn_s_wait_tensorcnt)
#define HAVE_TDM 1
#endif
#endif

union FragA { v16bf v; u16 u[16]; unsigned d[8]; };
union FragC { v8f v; float f[8]; };

__device__ __forceinline__ u16 f2bf(float f){
  union { float f; unsigned u; } x; x.f = f;
  unsigned r = x.u + 0x7FFFu + ((x.u >> 16) & 1u);
  return (u16)(r >> 16);
}
__device__ __forceinline__ int kmap(int i, int hi){ return (i < 8 ? i : i + 8) + hi*8; }

#if HAVE_TDM
// Issue one 128x32 bf16 tile (line stride K elements) -> LDS at ldsOff.
// HW pad: 16 DWORDs interval + 4 DWORDs pad => 80-byte LDS row stride.
__device__ __forceinline__ void tdm_load_tile(const u16* gsrc, unsigned ldsOff, int K){
  unsigned long long ga = (unsigned long long)gsrc;
  u32x4 g0;
  g0.x = 1u;                                            // count=1, no gather
  g0.y = ldsOff;                                        // lds_addr
  g0.z = (unsigned)(ga & 0xFFFFFFFFu);                  // global_addr[31:0]
  g0.w = (unsigned)((ga >> 32) & 0x01FFFFFFu) | 0x80000000u; // [56:32] | type=2
  i32x8 g1;
  g1[0] = (1 << 16) | (1 << 20) | (3 << 22) | (3 << 25); // 2B, pad_en, 16dw/4dw
  g1[1] = (int)((unsigned)(K & 0xFFFF) << 16);           // tensor_dim0 lo
  g1[2] = (int)(((unsigned)K >> 16) | ((unsigned)(MROWS & 0xFFFF) << 16));
  g1[3] = (32 << 16);                                    // tile_dim0 = 32
  g1[4] = 128;                                           // tile_dim1 = 128
  g1[5] = K;                                             // dim0 stride
  g1[6] = 0; g1[7] = 0;
  i32x4 gz4 = {0,0,0,0};
  i32x8 gz8 = {0,0,0,0,0,0,0,0};
  __builtin_amdgcn_tensor_load_to_lds(g0, g1, gz4, gz4, gz8, 0);
}
#endif

// ---------------------------------------------------------------------------
// x + pos_emb  ->  xc (f32)
// ---------------------------------------------------------------------------
__global__ void k_addpos(const float* __restrict__ x, const float* __restrict__ pe,
                         float* __restrict__ xc, int n){
  int i = blockIdx.x*256 + threadIdx.x;
  if (i < n) xc[i] = x[i] + pe[i % (TT*DIMM)];
}

// ---------------------------------------------------------------------------
// LayerNorm over DIM=768; output bf16 (for GEMM input) or f32 (final output)
// ---------------------------------------------------------------------------
template<bool OBF16>
__global__ void __launch_bounds__(256) k_ln(const float* __restrict__ x,
                                            const float* __restrict__ g,
                                            const float* __restrict__ b,
                                            void* __restrict__ outp){
  __shared__ float red[256];
  const int row = blockIdx.x, t = threadIdx.x;
  const float* xr = x + (size_t)row * DIMM;
  float lv[3]; float s = 0.f;
  #pragma unroll
  for (int j = 0; j < 3; ++j){ lv[j] = xr[t + j*256]; s += lv[j]; }
  red[t] = s; __syncthreads();
  for (int o = 128; o > 0; o >>= 1){ if (t < o) red[t] += red[t+o]; __syncthreads(); }
  const float mean = red[0] * (1.0f/DIMM);
  __syncthreads();
  float vs = 0.f;
  #pragma unroll
  for (int j = 0; j < 3; ++j){ float d = lv[j]-mean; vs += d*d; }
  red[t] = vs; __syncthreads();
  for (int o = 128; o > 0; o >>= 1){ if (t < o) red[t] += red[t+o]; __syncthreads(); }
  const float inv = rsqrtf(red[0]*(1.0f/DIMM) + 1e-5f);
  u16* ob = (u16*)outp; float* of = (float*)outp;
  #pragma unroll
  for (int j = 0; j < 3; ++j){
    int c = t + j*256;
    float vv = (lv[j]-mean)*inv*g[c] + b[c];
    if (OBF16) ob[(size_t)row*DIMM + c] = f2bf(vv);
    else       of[(size_t)row*DIMM + c] = vv;
  }
}

// ---------------------------------------------------------------------------
// GEMM: C[M,N] = epilogue( A_bf16[M,K] x W_f32->bf16[K,N] )
// 128x128 tile / 256 threads (8 waves). Software-pipelined:
//   - A tiles streamed by the Tensor Data Mover into double-buffered LDS
//     (issue TDM(i+1) while WMMAs consume tile i; s_wait_tensorcnt gates).
//   - B (fp32 weights) prefetched into registers one K-step ahead,
//     converted fp32->bf16 on LDS store.
// ---------------------------------------------------------------------------
template<bool BIAS, bool GELU, bool RES, bool OBF16>
__global__ void __launch_bounds__(256) k_gemm(const u16* __restrict__ A,
                                              const float* __restrict__ W,
                                              const float* __restrict__ bias,
                                              const float* __restrict__ res,
                                              void* __restrict__ outp,
                                              int K, int N){
  extern __shared__ char smem[];                   // no static LDS (TDM lds_addr)
  u16 (*As)[40] = (u16(*)[40])smem;                // 2 x [128][40]  (20480 B)
  u16 (*Bs)[40] = (u16(*)[40])(smem + 20480);      // 2 x [128][40]  (20480 B)
  const int t = threadIdx.x, lane = t & 31, wave = t >> 5;
  const int hi = (lane >> 4) & 1, ln = lane & 15;
  const int rowT = blockIdx.x * 128, colT = blockIdx.y * 128;
  FragC acc[8] = {};
  const int steps = K >> 5;

  // ---- prologue: B(0) into regs, A(0) -> As slot 0 ----
  float4 breg[4];
  #pragma unroll
  for (int q = 0; q < 4; ++q) {
    int fidx = t*4 + q;
    int kk = fidx >> 5, n4 = (fidx & 31) * 4;
    breg[q] = *(const float4*)(W + (size_t)kk*N + colT + n4);
  }
#if HAVE_TDM
  if (wave == 0) tdm_load_tile(A + (size_t)rowT*K, 0u, K);
#else
  { const int ra = t >> 1, sega = (t & 1) * 16;
    const uint4* ap = (const uint4*)(A + (size_t)(rowT + ra)*K + sega);
    uint4 a0 = ap[0], a1 = ap[1];
    *(uint4*)&As[ra][sega]     = a0;
    *(uint4*)&As[ra][sega + 8] = a1; }
#endif

  for (int i = 0; i < steps; ++i) {
    const int buf = i & 1;
    u16 (*Asb)[40] = As + buf*128;
    u16 (*Bsb)[40] = Bs + buf*128;

    // ---- publish B(i): regs -> LDS (fp32 -> bf16, transposed [n][k]) ----
    #pragma unroll
    for (int q = 0; q < 4; ++q) {
      int fidx = t*4 + q;
      int kk = fidx >> 5, n4 = (fidx & 31) * 4;
      Bsb[n4+0][kk] = f2bf(breg[q].x);
      Bsb[n4+1][kk] = f2bf(breg[q].y);
      Bsb[n4+2][kk] = f2bf(breg[q].z);
      Bsb[n4+3][kk] = f2bf(breg[q].w);
    }
    __syncthreads();                    // all compute(i-1) done; Bs[i&1] staged
#if HAVE_TDM
    if (wave == 0) __builtin_amdgcn_s_wait_tensorcnt(0);   // TDM(i) complete
#endif
    __syncthreads();                    // publish As[i&1] to all waves

    // ---- kick off next tile's traffic (overlaps the WMMAs below) ----
    if (i + 1 < steps) {
#if HAVE_TDM
      if (wave == 0)
        tdm_load_tile(A + (size_t)rowT*K + (i+1)*32, (unsigned)(((i+1)&1)*10240), K);
#else
      { const int ra = t >> 1, sega = (t & 1) * 16;
        u16 (*Asn)[40] = As + ((i+1)&1)*128;
        const uint4* ap = (const uint4*)(A + (size_t)(rowT + ra)*K + (i+1)*32 + sega);
        uint4 a0 = ap[0], a1 = ap[1];
        *(uint4*)&Asn[ra][sega]     = a0;
        *(uint4*)&Asn[ra][sega + 8] = a1; }
#endif
      #pragma unroll
      for (int q = 0; q < 4; ++q) {
        int fidx = t*4 + q;
        int kk = fidx >> 5, n4 = (fidx & 31) * 4;
        breg[q] = *(const float4*)(W + (size_t)((i+1)*32 + kk)*N + colT + n4);
      }
    }

    // ---- fragments (packed b32 pair reads) + 8 WMMAs ----
    FragA af;
    #pragma unroll
    for (int j = 0; j < 8; ++j)
      af.d[j] = *(const unsigned*)&Asb[wave*16 + ln][kmap(2*j, hi)];
    #pragma unroll
    for (int tn = 0; tn < 8; ++tn) {
      FragA bfr;
      #pragma unroll
      for (int j = 0; j < 8; ++j)
        bfr.d[j] = *(const unsigned*)&Bsb[tn*16 + ln][kmap(2*j, hi)];
      acc[tn].v = __builtin_amdgcn_wmma_f32_16x16x32_bf16(
          false, af.v, false, bfr.v, (short)0, acc[tn].v, false, false);
    }
  }

  // ---- epilogue ----
  u16* ob = (u16*)outp; float* of = (float*)outp;
  const int row0 = rowT + wave*16 + hi*8;
  #pragma unroll
  for (int tn = 0; tn < 8; ++tn) {
    const int col = colT + tn*16 + ln;
    #pragma unroll
    for (int r = 0; r < 8; ++r) {
      int row = row0 + r;
      float vv = acc[tn].f[r];
      if (BIAS) vv += bias[col];
      if (GELU) vv = 0.5f * vv * (1.0f + erff(vv * 0.70710678118f));
      if (RES)  vv += res[(size_t)row*N + col];
      if (OBF16) ob[(size_t)row*N + col] = f2bf(vv);
      else       of[(size_t)row*N + col] = vv;
    }
  }
}

// ---------------------------------------------------------------------------
// Flash attention with causal mask + softmaxed lag bias (bias table in LDS,
// dynamic-indexed ds_load). Block = 128 threads (4 waves); wave handles one
// 16-row tile of one (b,h); 32 keys per iteration, online softmax.
// ---------------------------------------------------------------------------
__global__ void __launch_bounds__(128) k_attn(const u16* __restrict__ q,
                                              const u16* __restrict__ k,
                                              const u16* __restrict__ v,
                                              const float* __restrict__ lw,
                                              u16* __restrict__ ao){
  __shared__ u16 Pl[4][16][40];
  __shared__ float lps[6];
  const int t = threadIdx.x, lane = t & 31, wave = t >> 5;
  const int hi = (lane >> 4) & 1, ln = lane & 15;
  const int tile = blockIdx.x*4 + wave;
  const int rows0 = tile*16;
  const int h = blockIdx.y, bb = blockIdx.z;

  if (t == 0) {               // lag-bias softmax (6 values, per head) -> LDS
    float lv[6]; float mx = -1e30f;
    #pragma unroll
    for (int i = 0; i < 6; ++i){ lv[i] = lw[h*6 + i]; mx = fmaxf(mx, lv[i]); }
    float sm = 0.f;
    #pragma unroll
    for (int i = 0; i < 6; ++i){ lv[i] = __expf(lv[i]-mx); sm += lv[i]; }
    float rs = 1.f/sm;
    #pragma unroll
    for (int i = 0; i < 6; ++i) lps[i] = lv[i]*rs;
  }
  __syncthreads();

  const size_t baseQ = ((size_t)bb*TT)*INNERN + (size_t)h*DHH;

  // Q fragments (16 rows x 64 feat = 2 A-frags), packed b32 reads
  FragA qa[2];
  { const u16* qp = q + baseQ + (size_t)(rows0 + ln)*INNERN;
    #pragma unroll
    for (int f = 0; f < 2; ++f)
      #pragma unroll
      for (int j = 0; j < 8; ++j)
        qa[f].d[j] = *(const unsigned*)(qp + f*32 + kmap(2*j, hi));
  }

  FragC o4[4] = {};
  float m[8], l[8];
  #pragma unroll
  for (int r = 0; r < 8; ++r){ m[r] = -1e30f; l[r] = 0.f; }
  const float scale = 0.125f;   // 1/sqrt(64)

  for (int jb = 0; jb < rows0 + 16; jb += 32) {
    // ---- S = Q K^T for keys [jb, jb+32) ----
    FragC s0 = {}, s1 = {};
    { const u16* kp0 = k + baseQ + (size_t)(jb + ln)*INNERN;
      const u16* kp1 = k + baseQ + (size_t)(jb + 16 + ln)*INNERN;
      #pragma unroll
      for (int f = 0; f < 2; ++f) {
        FragA kb0, kb1;
        #pragma unroll
        for (int j = 0; j < 8; ++j){
          int kk = f*32 + kmap(2*j, hi);
          kb0.d[j] = *(const unsigned*)(kp0 + kk);
          kb1.d[j] = *(const unsigned*)(kp1 + kk);
        }
        s0.v = __builtin_amdgcn_wmma_f32_16x16x32_bf16(
            false, qa[f].v, false, kb0.v, (short)0, s0.v, false, false);
        s1.v = __builtin_amdgcn_wmma_f32_16x16x32_bf16(
            false, qa[f].v, false, kb1.v, (short)0, s1.v, false, false);
      }
    }
    // ---- mask + lag bias + online softmax (rows across 16-lane groups) ----
    #pragma unroll
    for (int r = 0; r < 8; ++r) {
      const int row = rows0 + r + hi*8;
      const int j0 = jb + ln, j1 = j0 + 16;
      const int d0 = row - j0, d1 = row - j1;
      const int i0 = d0 < 0 ? 0 : (d0 > 5 ? 5 : d0);
      const int i1 = d1 < 0 ? 0 : (d1 > 5 ? 5 : d1);
      const float b0 = lps[i0], b1 = lps[i1];      // ds_load, dynamic index
      float sc0 = (d0 >= 0) ? s0.f[r]*scale + b0 : -1e30f;
      float sc1 = (d1 >= 0) ? s1.f[r]*scale + b1 : -1e30f;
      float rm = fmaxf(sc0, sc1);
      #pragma unroll
      for (int off = 8; off > 0; off >>= 1) rm = fmaxf(rm, __shfl_xor(rm, off, 16));
      float mn = fmaxf(m[r], rm);
      float corr = __expf(m[r] - mn);
      float p0 = __expf(sc0 - mn);
      float p1 = __expf(sc1 - mn);
      float rs = p0 + p1;
      #pragma unroll
      for (int off = 8; off > 0; off >>= 1) rs += __shfl_xor(rs, off, 16);
      l[r] = l[r]*corr + rs;
      m[r] = mn;
      #pragma unroll
      for (int tt = 0; tt < 4; ++tt) o4[tt].f[r] *= corr;
      Pl[wave][r + hi*8][ln]      = f2bf(p0);
      Pl[wave][r + hi*8][ln + 16] = f2bf(p1);
    }
    asm volatile("s_wait_dscnt 0" ::: "memory");   // D-layout -> A-layout via LDS
    FragA pa;
    #pragma unroll
    for (int j = 0; j < 8; ++j)
      pa.d[j] = *(const unsigned*)&Pl[wave][ln][kmap(2*j, hi)];
    // ---- O += P x V ----
    #pragma unroll
    for (int tt = 0; tt < 4; ++tt) {
      FragA vbf;
      #pragma unroll
      for (int i = 0; i < 16; ++i)
        vbf.u[i] = v[baseQ + (size_t)(jb + kmap(i, hi))*INNERN + tt*16 + ln];
      o4[tt].v = __builtin_amdgcn_wmma_f32_16x16x32_bf16(
          false, pa.v, false, vbf.v, (short)0, o4[tt].v, false, false);
    }
  }

  // ---- normalize & store (bf16) ----
  #pragma unroll
  for (int tt = 0; tt < 4; ++tt)
    #pragma unroll
    for (int r = 0; r < 8; ++r) {
      int row = rows0 + r + hi*8;
      float ov = o4[tt].f[r] / l[r];
      ao[baseQ + (size_t)row*INNERN + tt*16 + ln] = f2bf(ov);
    }
}

// ---------------------------------------------------------------------------
extern "C" void kernel_launch(void* const* d_in, const int* in_sizes, int n_in,
                              void* d_out, int out_size, void* d_ws, size_t ws_size,
                              hipStream_t stream){
  (void)in_sizes; (void)n_in; (void)out_size; (void)ws_size;
  const float* x    = (const float*)d_in[0];
  const float* pos  = (const float*)d_in[1];
  const float* ln1w = (const float*)d_in[2];
  const float* ln1b = (const float*)d_in[3];
  const float* wq   = (const float*)d_in[4];
  const float* wk   = (const float*)d_in[5];
  const float* wv   = (const float*)d_in[6];
  const float* lagw = (const float*)d_in[7];
  const float* wo   = (const float*)d_in[8];
  const float* wob  = (const float*)d_in[9];
  const float* ln2w = (const float*)d_in[10];
  const float* ln2b = (const float*)d_in[11];
  const float* w1   = (const float*)d_in[12];
  const float* b1   = (const float*)d_in[13];
  const float* w2   = (const float*)d_in[14];
  const float* b2   = (const float*)d_in[15];
  const float* lnfw = (const float*)d_in[16];
  const float* lnfb = (const float*)d_in[17];
  float* out = (float*)d_out;

  const size_t MT = (size_t)BB*TT;
  float* xc = (float*)d_ws;                 // 2048x768 f32 residual stream
  u16* xn = (u16*)(xc + MT*DIMM);           // 2048x768 bf16 (ln output)
  u16* qb = xn + MT*DIMM;                   // q/k/v bf16
  u16* kb = qb + MT*DIMM;
  u16* vb = kb + MT*DIMM;
  u16* ao = vb + MT*DIMM;                   // attention out bf16
  u16* hb = ao + MT*DIMM;                   // 2048x3072 bf16 (ffn hidden)

  { int n = (int)(MT*DIMM);
    k_addpos<<<dim3((n+255)/256), dim3(256), 0, stream>>>(x, pos, xc, n); }

  const dim3 g66(16, 6), g624(16, 24);
  const size_t GSM = 40960;                 // dynamic LDS: 2xAs + 2xBs
  for (int L = 0; L < DEPTHN; ++L) {
    const float* Wq = wq + (size_t)L*DIMM*INNERN;
    const float* Wk = wk + (size_t)L*DIMM*INNERN;
    const float* Wv = wv + (size_t)L*DIMM*INNERN;
    const float* Wo = wo + (size_t)L*INNERN*DIMM;
    const float* W1 = w1 + (size_t)L*DIMM*FFQ;
    const float* W2 = w2 + (size_t)L*FFQ*DIMM;

    k_ln<true><<<dim3(MROWS), 256, 0, stream>>>(xc, ln1w + L*DIMM, ln1b + L*DIMM, xn);
    k_gemm<false,false,false,true><<<g66, 256, GSM, stream>>>(xn, Wq, nullptr, nullptr, qb, DIMM, INNERN);
    k_gemm<false,false,false,true><<<g66, 256, GSM, stream>>>(xn, Wk, nullptr, nullptr, kb, DIMM, INNERN);
    k_gemm<false,false,false,true><<<g66, 256, GSM, stream>>>(xn, Wv, nullptr, nullptr, vb, DIMM, INNERN);
    k_attn<<<dim3(TT/64, HEADSN, BB), 128, 0, stream>>>(qb, kb, vb, lagw + L*HEADSN*6, ao);
    k_gemm<true,false,true,false><<<g66, 256, GSM, stream>>>(ao, Wo, wob + L*DIMM, xc, xc, INNERN, DIMM);
    k_ln<true><<<dim3(MROWS), 256, 0, stream>>>(xc, ln2w + L*DIMM, ln2b + L*DIMM, xn);
    k_gemm<true,true,false,true><<<g624, 256, GSM, stream>>>(xn, W1, b1 + L*FFQ, nullptr, hb, DIMM, FFQ);
    k_gemm<true,false,true,false><<<g66, 256, GSM, stream>>>(hb, W2, b2 + L*DIMM, xc, xc, FFQ, DIMM);
  }
  k_ln<false><<<dim3(MROWS), 256, 0, stream>>>(xc, lnfw, lnfb, out);
}
